// ChildSumTreeRNN_41867341201796
// MI455X (gfx1250) — compile-verified, matrix operand order checked by reading, not run on hardware
//
#include <hip/hip_runtime.h>
#include <hip/hip_bf16.h>
#include <math.h>

typedef __bf16 bf16_t;
typedef __attribute__((ext_vector_type(16))) __bf16 v16bf;
typedef __attribute__((ext_vector_type(8)))  __bf16 v8bf;
typedef __attribute__((ext_vector_type(8)))  float  v8f;

#define BATCH 1024
#define LEAVES 128
#define MDIM 256

// ---------------------------------------------------------------------------
// Hardware tanh (gfx1250 TRANS pipe) — avoids the ~20-op libm expansion.
// ---------------------------------------------------------------------------
__device__ __forceinline__ float hw_tanh(float x) {
#if __has_builtin(__builtin_amdgcn_tanhf)
  return __builtin_amdgcn_tanhf(x);
#elif __has_builtin(__builtin_amdgcn_tanh_f32)
  return __builtin_amdgcn_tanh_f32(x);
#else
  float y;
  asm volatile("v_tanh_f32 %0, %1" : "=v"(y) : "v"(x));
  return y;
#endif
}

// ---------------------------------------------------------------------------
// f32 -> bf16 conversion (weights + leaf inputs)
// ---------------------------------------------------------------------------
__global__ void __launch_bounds__(256) cvt_f32_bf16_kernel(
    const float* __restrict__ in, bf16_t* __restrict__ out, int n) {
  int i = blockIdx.x * 256 + threadIdx.x;
  if (i < n) out[i] = (bf16_t)in[i];
}

// Load one lane's slice of a 16x32 bf16 tile in WMMA A/B register order:
// elements 0..7  = K[khalf .. khalf+7]      (contiguous, 16 bytes)
// elements 8..15 = K[khalf+16 .. khalf+23]  (contiguous, 16 bytes)
__device__ __forceinline__ v16bf load_tile16(const bf16_t* __restrict__ p) {
  v8bf lo = *(const v8bf*)(p);
  v8bf hi = *(const v8bf*)(p + 16);
  v16bf r;
#pragma unroll
  for (int i = 0; i < 8; ++i) { r[i] = lo[i]; r[i + 8] = hi[i]; }
  return r;
}

// ---------------------------------------------------------------------------
// One tree level:  Y[parent] = mean_pair( tanh( X @ W_sel^T ) )
// X: (B*Ln) x 256 bf16 child states, row-major.
// Wa/Wo: 256x256 bf16 (row k = output feature, contiguous along m).
// Block: 256 threads = 8 waves; block handles 32 child rows x all 256 cols.
// Wave w owns column tiles n0=32w..32w+31 and BOTH 16-row tiles:
//   8 accumulators = 2 row tiles x 2 col tiles x 2 weights.
// Per K-step: 2 A tiles + 4 B tiles (12 b128 loads) feed 8 WMMAs.
// ---------------------------------------------------------------------------
__global__ void __launch_bounds__(256) tree_level_kernel(
    const bf16_t* __restrict__ X,
    const bf16_t* __restrict__ Wa,
    const bf16_t* __restrict__ Wo,
    const int*    __restrict__ ops,
    bf16_t* __restrict__ Yb,   // bf16 output (intermediate levels)
    float*  __restrict__ Yf,   // f32 output (final level)
    int logLn,                 // log2(children per batch at this level)
    int opOff,                 // offset into ops[:, off .. off+n)
    int finalLevel) {
  const int lane    = threadIdx.x & 31;
  const int wave    = threadIdx.x >> 5;
  const int rowBase = blockIdx.x * 32;          // 2 x 16 child-row tiles
  const int n0      = wave * 32;                // two 16-wide col tiles
  const int rsel    = lane & 15;
  const int khalf   = (lane >> 4) * 8;          // 16-bit A/B layout half-split

  const bf16_t* aP0 = X  + (size_t)(rowBase +      rsel) * MDIM + khalf;
  const bf16_t* aP1 = X  + (size_t)(rowBase + 16 + rsel) * MDIM + khalf;
  const bf16_t* bA0 = Wa + (size_t)(n0 +      rsel) * MDIM + khalf;
  const bf16_t* bA1 = Wa + (size_t)(n0 + 16 + rsel) * MDIM + khalf;
  const bf16_t* bO0 = Wo + (size_t)(n0 +      rsel) * MDIM + khalf;
  const bf16_t* bO1 = Wo + (size_t)(n0 + 16 + rsel) * MDIM + khalf;

  // acc[rowTile][mat*2+colTile]: mat 0 = W_and, 1 = W_or
  v8f aa0 = {}, aa1 = {}, ao0 = {}, ao1 = {};   // row tile 0
  v8f ba0 = {}, ba1 = {}, bo0 = {}, bo1 = {};   // row tile 1

#pragma unroll
  for (int kk = 0; kk < MDIM / 32; ++kk) {
    const int kb = kk * 32;
    v16bf a0 = load_tile16(aP0 + kb);
    v16bf a1 = load_tile16(aP1 + kb);
    v16bf w0 = load_tile16(bA0 + kb);
    aa0 = __builtin_amdgcn_wmma_f32_16x16x32_bf16(false, a0, false, w0,
                                                  (short)0, aa0, false, false);
    ba0 = __builtin_amdgcn_wmma_f32_16x16x32_bf16(false, a1, false, w0,
                                                  (short)0, ba0, false, false);
    v16bf w1 = load_tile16(bA1 + kb);
    aa1 = __builtin_amdgcn_wmma_f32_16x16x32_bf16(false, a0, false, w1,
                                                  (short)0, aa1, false, false);
    ba1 = __builtin_amdgcn_wmma_f32_16x16x32_bf16(false, a1, false, w1,
                                                  (short)0, ba1, false, false);
    v16bf w2 = load_tile16(bO0 + kb);
    ao0 = __builtin_amdgcn_wmma_f32_16x16x32_bf16(false, a0, false, w2,
                                                  (short)0, ao0, false, false);
    bo0 = __builtin_amdgcn_wmma_f32_16x16x32_bf16(false, a1, false, w2,
                                                  (short)0, bo0, false, false);
    v16bf w3 = load_tile16(bO1 + kb);
    ao1 = __builtin_amdgcn_wmma_f32_16x16x32_bf16(false, a0, false, w3,
                                                  (short)0, ao1, false, false);
    bo1 = __builtin_amdgcn_wmma_f32_16x16x32_bf16(false, a1, false, w3,
                                                  (short)0, bo1, false, false);
  }

  // tanh on all 64 accumulated values — one TRANS instruction each.
#pragma unroll
  for (int i = 0; i < 8; ++i) {
    aa0[i] = hw_tanh(aa0[i]); aa1[i] = hw_tanh(aa1[i]);
    ao0[i] = hw_tanh(ao0[i]); ao1[i] = hw_tanh(ao1[i]);
    ba0[i] = hw_tanh(ba0[i]); ba1[i] = hw_tanh(ba1[i]);
    bo0[i] = hw_tanh(bo0[i]); bo1[i] = hw_tanh(bo1[i]);
  }

  // Sibling pair-mean in registers: C VGPR r holds row r (lanes 0-15)
  // or row r+8 (lanes 16-31); pairs are (r, r+1) for even r.
  const int Ln   = 1 << logLn;
  const int col0 = n0 + rsel;
  const int col1 = n0 + 16 + rsel;

#pragma unroll
  for (int t = 0; t < 2; ++t) {                     // row tile
    const v8f& ca0 = t ? ba0 : aa0;
    const v8f& ca1 = t ? ba1 : aa1;
    const v8f& co0 = t ? bo0 : ao0;
    const v8f& co1 = t ? bo1 : ao1;
    const int tileBase = rowBase + t * 16;
#pragma unroll
    for (int r = 0; r < 8; r += 2) {
      const int localM = r + ((lane >> 4) * 8);     // even child row in tile
      const int g  = tileBase + localM;             // global (even) child row
      const int b  = g >> logLn;                    // batch index
      const int p  = (g & (Ln - 1)) >> 1;           // pair index in batch
      const int op = ops[b * (LEAVES - 1) + opOff + p];
      const int orow = g >> 1;                      // parent flat row

      const float va0 = 0.5f * (ca0[r] + ca0[r + 1]);
      const float vo0 = 0.5f * (co0[r] + co0[r + 1]);
      const float va1 = 0.5f * (ca1[r] + ca1[r + 1]);
      const float vo1 = 0.5f * (co1[r] + co1[r + 1]);
      const float v0 = (op != 0) ? vo0 : va0;
      const float v1 = (op != 0) ? vo1 : va1;

      if (finalLevel) {
        Yf[(size_t)orow * MDIM + col0] = v0;
        Yf[(size_t)orow * MDIM + col1] = v1;
      } else {
        Yb[(size_t)orow * MDIM + col0] = (bf16_t)v0;
        Yb[(size_t)orow * MDIM + col1] = (bf16_t)v1;
      }
    }
  }
}

// ---------------------------------------------------------------------------
// Host-side orchestration
// ---------------------------------------------------------------------------
extern "C" void kernel_launch(void* const* d_in, const int* in_sizes, int n_in,
                              void* d_out, int out_size, void* d_ws, size_t ws_size,
                              hipStream_t stream) {
  const float* inputs = (const float*)d_in[0];   // (B, 128, 256) f32
  const int*   ops    = (const int*)d_in[1];     // (B, 127) i32
  const float* W_and  = (const float*)d_in[2];   // (256, 256) f32
  const float* W_or   = (const float*)d_in[3];   // (256, 256) f32
  float* out = (float*)d_out;                    // (B, 256) f32

  char* ws = (char*)d_ws;
  const size_t wElems  = (size_t)MDIM * MDIM;                   // 65536
  const size_t x0Elems = (size_t)BATCH * LEAVES * MDIM;         // 33.5M
  bf16_t* wa = (bf16_t*)ws;                                     // 128 KB
  bf16_t* wo = (bf16_t*)(ws + wElems * sizeof(bf16_t));         // 128 KB
  bf16_t* x0 = (bf16_t*)(ws + 2 * wElems * sizeof(bf16_t));     // 64 MB
  bf16_t* x1 = (bf16_t*)((char*)x0 + x0Elems * sizeof(bf16_t)); // 32 MB used

  // Convert weights + leaf states to bf16.
  cvt_f32_bf16_kernel<<<(int)((wElems + 255) / 256), 256, 0, stream>>>(
      W_and, wa, (int)wElems);
  cvt_f32_bf16_kernel<<<(int)((wElems + 255) / 256), 256, 0, stream>>>(
      W_or, wo, (int)wElems);
  cvt_f32_bf16_kernel<<<(int)((x0Elems + 255) / 256), 256, 0, stream>>>(
      inputs, x0, (int)x0Elems);

  // 7 levels, bottom-up; ping-pong h buffers.
  bf16_t* cur = x0;
  bf16_t* nxt = x1;
  int off = 0;
  for (int d = 0; d < 7; ++d) {
    const int Ln     = LEAVES >> d;       // children per batch this level
    const int logLn  = 7 - d;
    const int rows   = BATCH * Ln;        // total child rows (multiple of 32)
    const int blocks = rows / 32;         // 32 child rows per block
    const bool fin   = (d == 6);
    tree_level_kernel<<<blocks, 256, 0, stream>>>(
        cur, wa, wo, ops,
        fin ? (bf16_t*)nullptr : nxt,
        fin ? out : (float*)nullptr,
        logLn, off, fin ? 1 : 0);
    off += Ln / 2;
    bf16_t* t = cur; cur = nxt; nxt = t;
  }
}
